// rigid_mLSTM_45707041964057
// MI455X (gfx1250) — matrix operand order repaired and checked by reading, not compiled, not run on hardware
//
#include <hip/hip_runtime.h>
#include <hip/hip_bf16.h>

typedef __attribute__((ext_vector_type(16))) _Float16 v16h;
typedef __attribute__((ext_vector_type(8)))  float    v8f;

#define BATCH   8
#define TSTEPS  128
#define INDIM   32
#define HDIM    256
#define SIXH    1536
#define NTILES  96      // SIXH / 16
#define KS2     8       // HDIM / 32
#define NWAVES  8
#define NTHREADS 256

// ---------------------------------------------------------------------------
// Prep kernel: repack W1 (1536x32) and W2 (1536x256) fp32 row-major into f16,
// laid out in this file's WMMA B-fragment order (16-bit B 32x16, wave32, contiguous k as in A):
//   lane L holds column n = L&15 ; half h holds k = ((L<16)?0:16) + h
// Fragment index = ((tile*ksteps + ks)*32 + lane)*16 + h  -> one contiguous
// 32-byte v16h load per lane in the main loop.
// ---------------------------------------------------------------------------
__global__ void pack_weights_f16(const float* __restrict__ W1,
                                 const float* __restrict__ W2,
                                 _Float16* __restrict__ wb1,
                                 _Float16* __restrict__ wb2) {
    int idx = blockIdx.x * blockDim.x + threadIdx.x;
    const int N1 = NTILES * 32 * 16;          // W1 pack: 1 k-step
    const int N2 = NTILES * KS2 * 32 * 16;    // W2 pack: 8 k-steps
    if (idx < N1) {
        int h = idx & 15, lane = (idx >> 4) & 31, nt = idx >> 9;
        int n = nt * 16 + (lane & 15);
        int k = ((lane < 16) ? 0 : 16) + h;
        wb1[idx] = (_Float16)W1[n * INDIM + k];
    } else if (idx < N1 + N2) {
        int j = idx - N1;
        int h = j & 15, lane = (j >> 4) & 31, ks = (j >> 9) & 7, nt = j >> 12;
        int n = nt * 16 + (lane & 15);
        int k = ks * 32 + ((lane < 16) ? 0 : 16) + h;
        wb2[j] = (_Float16)W2[n * HDIM + k];
    }
}

// ---------------------------------------------------------------------------
// Gate GEMM: gates(16 x 1536) = A(16 x 32*KSTEPS) * Wb + bias, rows 0..7 valid.
// Each wave owns 12 N-tiles; accumulate in f32 via v_wmma_f32_16x16x32_f16.
// ---------------------------------------------------------------------------
template <int KSTEPS>
__device__ __forceinline__
void gate_gemm(const _Float16* __restrict__ wb, const float* __restrict__ bias,
               const v16h* afrag, float* __restrict__ gatesLds,
               int wave, int lane) {
    for (int tile = wave; tile < NTILES; tile += NWAVES) {
        float bval = bias[tile * 16 + (lane & 15)];
        v8f acc;
#pragma unroll
        for (int f = 0; f < 8; ++f) acc[f] = bval;

        const v16h* bp = ((const v16h*)wb) + (size_t)tile * KSTEPS * 32 + lane;
        if (tile + NWAVES < NTILES)   // warm L0/L2 for this wave's next tile
            __builtin_prefetch(((const v16h*)wb) +
                               (size_t)(tile + NWAVES) * KSTEPS * 32 + lane, 0, 1);
#pragma unroll
        for (int ks = 0; ks < KSTEPS; ++ks) {
            v16h bmat = bp[ks * 32];
            acc = __builtin_amdgcn_wmma_f32_16x16x32_f16(
                      false, afrag[ks], false, bmat, (short)0, acc, false, false);
        }
        if (lane < 16) {            // rows m=0..7 live in lanes 0-15, VGPR f=m
#pragma unroll
            for (int f = 0; f < 8; ++f)
                gatesLds[f * SIXH + tile * 16 + lane] = acc[f];
        }
    }
}

// ---------------------------------------------------------------------------
// mLSTM cell update (collapsed C: C[b,i,j] == c[b,i] since the source update
// broadcasts over j and C starts at 0).  Deterministic wave-shuffle reductions
// for Sq[b] = sum_j q[b,j] and Nq[b] = sum_i n_t[b,i] q[b,i].
// ---------------------------------------------------------------------------
__device__ __forceinline__
void cell_update(const float* __restrict__ g, float* c, float* n, float* m,
                 float* h, float* Sq, float* Nq, int tid, int wave, int lane) {
#pragma unroll
    for (int r = 0; r < (BATCH * HDIM) / NTHREADS; ++r) {
        int e = tid + r * NTHREADS;           // 0..2047
        int b = e >> 8, i = e & 255;
        const float* gb = g + b * SIXH;
        float ig = gb[i];
        float fg = gb[256 + i];
        float kk = gb[1024 + i] * 0.0625f;    // 1/sqrt(256)
        float vv = gb[1280 + i];
        float mp = m[e];
        float mt = fmaxf(fg + mp, ig);
        float it = expf(ig - mt);
        float ft = expf(fg + mp - mt);
        c[e] = ft * c[e] + it * vv * kk;
        n[e] = ft * n[e] + it * kk;
        m[e] = mt;
    }
    __syncthreads();
    {   // wave w reduces batch b = w  (8 waves == 8 batches)
        int b = wave;
        float sq = 0.f, snq = 0.f;
#pragma unroll
        for (int j = 0; j < HDIM / 32; ++j) {
            int i = lane + j * 32;
            float qv = g[b * SIXH + 768 + i];
            sq  += qv;
            snq += n[b * HDIM + i] * qv;
        }
#pragma unroll
        for (int off = 16; off > 0; off >>= 1) {
            sq  += __shfl_down(sq,  off, 32);
            snq += __shfl_down(snq, off, 32);
        }
        if (lane == 0) { Sq[b] = sq; Nq[b] = snq; }
    }
    __syncthreads();
#pragma unroll
    for (int r = 0; r < (BATCH * HDIM) / NTHREADS; ++r) {
        int e = tid + r * NTHREADS;
        int b = e >> 8, i = e & 255;
        float og  = g[b * SIXH + 512 + i];
        float sig = 1.f / (1.f + expf(-og));
        float den = fmaxf(fabsf(Nq[b]), 1e-6f);
        h[e] = sig * c[e] * Sq[b] / den;      // Cq = c * Sq (collapsed C)
    }
    __syncthreads();
}

// ---------------------------------------------------------------------------
// Persistent kernel: one workgroup (8 wave32s) runs the whole recurrence with
// all state in LDS, then the MLP head.
// ---------------------------------------------------------------------------
__global__ __launch_bounds__(NTHREADS)
void mlstm_persistent(const float* __restrict__ x,
                      const _Float16* __restrict__ wb1, const float* __restrict__ b1,
                      const _Float16* __restrict__ wb2, const float* __restrict__ b2,
                      const float* __restrict__ fc1_w, const float* __restrict__ fc1_b,
                      const float* __restrict__ fc2_w, const float* __restrict__ fc2_b,
                      const float* __restrict__ fc3_w, const float* __restrict__ fc3_b,
                      float* __restrict__ out) {
    __shared__ float gates[BATCH * SIXH];                         // 48 KB
    __shared__ float c1[BATCH * HDIM], n1[BATCH * HDIM],
                     m1[BATCH * HDIM], h1[BATCH * HDIM];          // 32 KB
    __shared__ float c2[BATCH * HDIM], n2[BATCH * HDIM],
                     m2[BATCH * HDIM], h2[BATCH * HDIM];          // 32 KB
    __shared__ float Sq[BATCH], Nq[BATCH];

    const int tid  = threadIdx.x;
    const int wave = tid >> 5;
    const int lane = tid & 31;
    const int mrow = lane & 15;                 // A-fragment row (batch)
    const int kg16 = (lane < 16) ? 0 : 16;      // A-fragment K base

    for (int i = tid; i < BATCH * HDIM; i += NTHREADS) {
        c1[i] = 0.f; n1[i] = 0.f; m1[i] = 0.f;
        c2[i] = 0.f; n2[i] = 0.f; m2[i] = 0.f;
    }
    __syncthreads();

    for (int t = 0; t < TSTEPS; ++t) {
        // ---- layer 1: A (16x32) from x[:, t, :] (rows >= 8 are zero pad)
        v16h a1[1];
#pragma unroll
        for (int hh = 0; hh < 16; ++hh) {
            int k = kg16 + hh;
            float v = (mrow < BATCH) ? x[(mrow * TSTEPS + t) * INDIM + k] : 0.f;
            a1[0][hh] = (_Float16)v;
        }
        gate_gemm<1>(wb1, b1, a1, gates, wave, lane);
        __syncthreads();
        cell_update(gates, c1, n1, m1, h1, Sq, Nq, tid, wave, lane);

        // ---- layer 2: A (16x256) from h1 in LDS
        v16h a2[KS2];
#pragma unroll
        for (int ks = 0; ks < KS2; ++ks)
#pragma unroll
            for (int hh = 0; hh < 16; ++hh) {
                int k = ks * 32 + kg16 + hh;
                float v = (mrow < BATCH) ? h1[mrow * HDIM + k] : 0.f;
                a2[ks][hh] = (_Float16)v;
            }
        gate_gemm<KS2>(wb2, b2, a2, gates, wave, lane);
        __syncthreads();
        cell_update(gates, c2, n2, m2, h2, Sq, Nq, tid, wave, lane);
    }

    // ---- MLP head on h2 (last timestep); reuse `gates` LDS for activations
    float* r1 = gates;                 // 8 x 128
    float* r2 = gates + BATCH * 128;   // 8 x 64
    for (int o = tid; o < BATCH * 128; o += NTHREADS) {
        int b = o >> 7, j = o & 127;
        float s = fc1_b[j];
        for (int i = 0; i < HDIM; ++i) s += h2[b * HDIM + i] * fc1_w[j * HDIM + i];
        r1[o] = fmaxf(s, 0.f);
    }
    __syncthreads();
    for (int o = tid; o < BATCH * 64; o += NTHREADS) {
        int b = o >> 6, j = o & 63;
        float s = fc2_b[j];
        for (int i = 0; i < 128; ++i) s += r1[b * 128 + i] * fc2_w[j * 128 + i];
        r2[o] = fmaxf(s, 0.f);
    }
    __syncthreads();
    if (tid < BATCH) {
        float s = fc3_b[0];
        for (int i = 0; i < 64; ++i) s += r2[tid * 64 + i] * fc3_w[i];
        out[tid] = s;
    }
}

// ---------------------------------------------------------------------------
extern "C" void kernel_launch(void* const* d_in, const int* in_sizes, int n_in,
                              void* d_out, int out_size, void* d_ws, size_t ws_size,
                              hipStream_t stream) {
    (void)in_sizes; (void)n_in; (void)out_size; (void)ws_size;
    const float* x     = (const float*)d_in[0];
    const float* W1    = (const float*)d_in[1];
    const float* b1    = (const float*)d_in[2];
    const float* W2    = (const float*)d_in[3];
    const float* b2    = (const float*)d_in[4];
    const float* fc1_w = (const float*)d_in[5];
    const float* fc1_b = (const float*)d_in[6];
    const float* fc2_w = (const float*)d_in[7];
    const float* fc2_b = (const float*)d_in[8];
    const float* fc3_w = (const float*)d_in[9];
    const float* fc3_b = (const float*)d_in[10];

    _Float16* wb1 = (_Float16*)d_ws;                    // 96 KB
    _Float16* wb2 = wb1 + NTILES * 32 * 16;             // 768 KB

    const int total = NTILES * 32 * 16 + NTILES * KS2 * 32 * 16;
    pack_weights_f16<<<(total + 255) / 256, 256, 0, stream>>>(W1, W2, wb1, wb2);
    mlstm_persistent<<<1, NTHREADS, 0, stream>>>(x, wb1, b1, wb2, b2,
        fc1_w, fc1_b, fc2_w, fc2_b, fc3_w, fc3_b, (float*)d_out);
}
